// VectorQuantizer_91061896610022
// MI455X (gfx1250) — matrix operand order repaired and checked by reading, not compiled, not run on hardware
//
#include <hip/hip_runtime.h>
#include <hip/hip_bf16.h>

typedef __attribute__((ext_vector_type(16))) _Float16 v16h;
typedef __attribute__((ext_vector_type(8)))  float    v8f;
typedef int v4i_ __attribute__((vector_size(16)));   // matches builtin param type

#define D_DIM   256
#define N_TOK   16384
#define K_CODES 16384
#define HW_     1024
#define CODE_TILE 64
#define EMB_SCALE 16384.0f
#define TWO_OVER_S (2.0f / EMB_SCALE)

#if defined(__has_builtin)
#if __has_builtin(__builtin_amdgcn_global_load_async_to_lds_b128)
#define HAVE_ASYNC_B128 1
#endif
#if __has_builtin(__builtin_amdgcn_s_wait_asynccnt)
#define HAVE_WAIT_ASYNC 1
#endif
#endif

__device__ __forceinline__ void async_wait0() {
#if defined(HAVE_WAIT_ASYNC)
    __builtin_amdgcn_s_wait_asynccnt(0);
#elif defined(HAVE_ASYNC_B128)
    asm volatile("s_wait_asynccnt 0x0" ::: "memory");
#endif
}

// ---------------------------------------------------------------- zero counts
__global__ void vq_zero_kernel(unsigned int* counts) {
    int i = blockIdx.x * blockDim.x + threadIdx.x;
    if (i < K_CODES) counts[i] = 0u;
}

// ----------------------------------------------- emb -> f16*S  +  ||e||^2 fp32
__global__ __launch_bounds__(256) void vq_prep_emb_kernel(
        const float* __restrict__ emb, _Float16* __restrict__ ef16,
        float* __restrict__ enorm) {
    const int code = blockIdx.x * 8 + (threadIdx.x >> 5);
    const int lane = threadIdx.x & 31;
    const float* row = emb + (size_t)code * D_DIM;
    _Float16* orow = ef16 + (size_t)code * D_DIM;
    float s = 0.0f;
#pragma unroll
    for (int i = 0; i < 8; ++i) {
        float v = row[lane + 32 * i];
        s += v * v;
        orow[lane + 32 * i] = (_Float16)(v * EMB_SCALE);
    }
#pragma unroll
    for (int m = 16; m >= 1; m >>= 1) s += __shfl_xor(s, m, 32);
    if (lane == 0) enorm[code] = s;
}

// ------------------------------------------------- fused GEMM + row-argmin
// grid = 128 blocks, block = 128 threads (4 waves). Wave w owns 32 tokens
// (two 16-row A slabs) so each LDS B fragment feeds TWO WMMAs.
// emb tiles double-buffered in LDS via async global->LDS copies.
__global__ __launch_bounds__(128) void vq_gemm_argmin_kernel(
        const float* __restrict__ z,        // NCHW fp32
        const _Float16* __restrict__ ef16,  // [K][256] f16 (scaled)
        const float* __restrict__ enorm,    // [K] fp32
        int* __restrict__ outidx) {
    __shared__ __align__(32) _Float16 etile[2][CODE_TILE * D_DIM];  // 2 x 32 KB
    __shared__ float enorm_t[2][CODE_TILE];

    const int tid    = threadIdx.x;
    const int wave   = tid >> 5;
    const int lane   = tid & 31;
    const int lane16 = lane & 15;
    const int hi16   = lane >> 4;           // 0 or 1
    const int tbase0 = blockIdx.x * 128 + wave * 32;
    const int tbase1 = tbase0 + 16;

    // ---- preload two A slabs (16x256 tokens each), reused for all codes
    v16h afrag[2][8];
#pragma unroll
    for (int s = 0; s < 2; ++s) {
        const int m  = (s ? tbase1 : tbase0) + lane16;
        const int b  = m >> 10;
        const int hw = m & 1023;
        const float* zrow = z + ((size_t)b * D_DIM) * HW_ + hw;
#pragma unroll
        for (int c = 0; c < 8; ++c) {
            v16h a;
#pragma unroll
            for (int j = 0; j < 8; ++j) {
                const int d0 = c * 32 + hi16 * 16 + j;      // K = hi*16 + j
                const int d1 = d0 + 8;                      // K = hi*16 + 8 + j
                a[j]     = (_Float16)zrow[(size_t)d0 * HW_];
                a[8 + j] = (_Float16)zrow[(size_t)d1 * HW_];
            }
            afrag[s][c] = a;
        }
    }

    float bestv[2][8];
    int   besti[2][8];
#pragma unroll
    for (int s = 0; s < 2; ++s)
#pragma unroll
        for (int j = 0; j < 8; ++j) { bestv[s][j] = 3.4e38f; besti[s][j] = 0; }

    // ---- tile stager: 64 codes x 256 halves = 2048 x 16B chunks
    auto stage = [&](int ct2, int nb) {
        const uint4* src = (const uint4*)(ef16 + (size_t)ct2 * D_DIM);
        uint4* dstl = (uint4*)etile[nb];
#if defined(HAVE_ASYNC_B128)
#pragma unroll
        for (int i = 0; i < 16; ++i) {
            const int c = i * 128 + tid;
            __builtin_amdgcn_global_load_async_to_lds_b128(
                (__attribute__((address_space(1))) v4i_*)(src + c),
                (__attribute__((address_space(3))) v4i_*)(dstl + c),
                0, 0);
        }
#else
#pragma unroll
        for (int i = 0; i < 16; ++i) {
            const int c = i * 128 + tid;
            dstl[c] = src[c];
        }
#endif
        if (tid < CODE_TILE) enorm_t[nb][tid] = enorm[ct2 + tid];
    };

    // prologue: stage tile 0
    stage(0, 0);
    async_wait0();
    __syncthreads();

    int buf = 0;
    for (int ct = 0; ct < K_CODES; ct += CODE_TILE) {
        const int nb = buf ^ 1;
        if (ct + CODE_TILE < K_CODES) {
            stage(ct + CODE_TILE, nb);
            __builtin_prefetch(ef16 + (size_t)(ct + 2 * CODE_TILE) * D_DIM, 0, 0);
        }

        // ---- 4 sub-tiles of 16 codes; each B fragment feeds 2 WMMAs
#pragma unroll
        for (int cs = 0; cs < 4; ++cs) {
            v8f acc0 = {};
            v8f acc1 = {};
            const _Float16* bp =
                &etile[buf][(size_t)(cs * 16 + lane16) * D_DIM + hi16 * 16];
#pragma unroll
            for (int dc = 0; dc < 8; ++dc) {
                v16h bf = *(const v16h*)(bp + dc * 32);
                acc0 = __builtin_amdgcn_wmma_f32_16x16x32_f16(
                        false, afrag[0][dc], false, bf, (short)0, acc0,
                        false, false);
                acc1 = __builtin_amdgcn_wmma_f32_16x16x32_f16(
                        false, afrag[1][dc], false, bf, (short)0, acc1,
                        false, false);
            }
            const float en = enorm_t[buf][cs * 16 + lane16];
            const int   ci = ct + cs * 16 + lane16;
#pragma unroll
            for (int j = 0; j < 8; ++j) {
                // score = ||e||^2 - 2 z.e   (||z||^2 constant per row)
                const float s0 = en - TWO_OVER_S * acc0[j];
                const float s1 = en - TWO_OVER_S * acc1[j];
                const bool l0 = s0 < bestv[0][j];
                const bool l1 = s1 < bestv[1][j];
                bestv[0][j] = l0 ? s0 : bestv[0][j];
                besti[0][j] = l0 ? ci : besti[0][j];
                bestv[1][j] = l1 ? s1 : bestv[1][j];
                besti[1][j] = l1 ? ci : besti[1][j];
            }
        }
        async_wait0();
        __syncthreads();
        buf = nb;
    }

    // ---- reduce argmin across the 16 lane-columns (ties -> lowest index)
#pragma unroll
    for (int s = 0; s < 2; ++s) {
#pragma unroll
        for (int mask = 1; mask <= 8; mask <<= 1) {
#pragma unroll
            for (int j = 0; j < 8; ++j) {
                const float ov = __shfl_xor(bestv[s][j], mask, 32);
                const int   oi = __shfl_xor(besti[s][j], mask, 32);
                if (ov < bestv[s][j] || (ov == bestv[s][j] && oi < besti[s][j])) {
                    bestv[s][j] = ov; besti[s][j] = oi;
                }
            }
        }
        if (lane16 == 0) {
            const int mrow = (s ? tbase1 : tbase0) + hi16 * 8;
#pragma unroll
            for (int j = 0; j < 8; ++j) outidx[mrow + j] = besti[s][j];
        }
    }
}

// ------------------------- gather z_q, NCHW output, loss partials, histogram
__global__ __launch_bounds__(256) void vq_gather_kernel(
        const float* __restrict__ z, const float* __restrict__ emb,
        const int* __restrict__ idx, float* __restrict__ out,
        float* __restrict__ outIdxF, float* __restrict__ lossPartial,
        unsigned int* __restrict__ counts) {
    __shared__ float wsum[8];
    const int n = blockIdx.x;
    const int d = threadIdx.x;
    const int k = idx[n];
    const float q = emb[(size_t)k * D_DIM + d];
    const int b = n >> 10, hw = n & 1023;
    const size_t off = ((size_t)b * D_DIM + d) * HW_ + hw;
    out[off] = q;                       // straight-through value == z_q
    const float diff = q - z[off];
    float v = diff * diff;
#pragma unroll
    for (int m = 16; m >= 1; m >>= 1) v += __shfl_xor(v, m, 32);
    if ((d & 31) == 0) wsum[d >> 5] = v;
    __syncthreads();
    if (d == 0) {
        float t = 0.0f;
#pragma unroll
        for (int w = 0; w < 8; ++w) t += wsum[w];   // fixed order -> deterministic
        lossPartial[n] = t;
        atomicAdd(&counts[k], 1u);                  // integer: deterministic
        outIdxF[n] = (float)k;
    }
}

// ------------------------------------------------ loss + perplexity finalize
__global__ __launch_bounds__(256) void vq_final_kernel(
        const unsigned int* __restrict__ counts,
        const float* __restrict__ lossPartial,
        float* __restrict__ outLoss, float* __restrict__ outPerp) {
    __shared__ float redH[8], redL[8];
    float h = 0.0f, l = 0.0f;
    for (int kk = threadIdx.x; kk < K_CODES; kk += 256) {
        const float p = (float)counts[kk] * (1.0f / (float)N_TOK);
        h += p * __logf(p + 1e-10f);
        l += lossPartial[kk];           // N_TOK == K_CODES: same stride walk
    }
#pragma unroll
    for (int m = 16; m >= 1; m >>= 1) {
        h += __shfl_xor(h, m, 32);
        l += __shfl_xor(l, m, 32);
    }
    if ((threadIdx.x & 31) == 0) { redH[threadIdx.x >> 5] = h; redL[threadIdx.x >> 5] = l; }
    __syncthreads();
    if (threadIdx.x == 0) {
        float th = 0.0f, tl = 0.0f;
#pragma unroll
        for (int w = 0; w < 8; ++w) { th += redH[w]; tl += redL[w]; }
        *outPerp = __expf(-th);
        *outLoss = 1.25f * tl * (1.0f / 4194304.0f);   // (1+BETA) * MSE
    }
}

// ---------------------------------------------------------------------------
extern "C" void kernel_launch(void* const* d_in, const int* in_sizes, int n_in,
                              void* d_out, int out_size, void* d_ws, size_t ws_size,
                              hipStream_t stream) {
    const float* z   = (const float*)d_in[0];   // [16,256,32,32]
    const float* emb = (const float*)d_in[1];   // [16384,256]

    float* out     = (float*)d_out;             // z_q NCHW
    float* outLoss = out + 4194304;
    float* outPerp = out + 4194305;
    float* outIdxF = out + 4194306;             // idx as float

    char* ws = (char*)d_ws;
    _Float16*     ef16        = (_Float16*)ws;                       // 8 MB
    float*        enorm       = (float*)(ws + 8388608);              // 64 KB
    int*          idx         = (int*)(ws + 8388608 + 65536);        // 64 KB
    unsigned int* counts      = (unsigned int*)(ws + 8388608 + 2 * 65536);
    float*        lossPartial = (float*)(ws + 8388608 + 3 * 65536);  // 64 KB

    vq_zero_kernel<<<64, 256, 0, stream>>>(counts);
    vq_prep_emb_kernel<<<K_CODES / 8, 256, 0, stream>>>(emb, ef16, enorm);
    vq_gemm_argmin_kernel<<<N_TOK / 128, 128, 0, stream>>>(z, ef16, enorm, idx);
    vq_gather_kernel<<<N_TOK, 256, 0, stream>>>(z, emb, idx, out, outIdxF,
                                                lossPartial, counts);
    vq_final_kernel<<<1, 256, 0, stream>>>(counts, lossPartial, outLoss, outPerp);
}
